// SSA_62938450756070
// MI455X (gfx1250) — matrix operand order repaired and checked by reading, not compiled, not run on hardware
//
#include <hip/hip_runtime.h>

#define TT 4
#define BB 32
#define NN 256
#define CC 512
#define HH 8
#define DD 64
#define MM (BB*NN)          // 8192 rows per time step
#define MC (MM*CC)          // 4194304 elements per time step

typedef __attribute__((ext_vector_type(16))) _Float16 v16h;
typedef __attribute__((ext_vector_type(8)))  _Float16 v8h;
typedef __attribute__((ext_vector_type(4)))  _Float16 v4h;
typedef __attribute__((ext_vector_type(8)))  float    v8f;
typedef __attribute__((ext_vector_type(4)))  float    v4f;
typedef __attribute__((ext_vector_type(4)))  unsigned int v4u;
typedef __attribute__((ext_vector_type(4)))  int      v4i;
typedef __attribute__((ext_vector_type(8)))  int      v8i;

__device__ __forceinline__ v8f zero8() {
  v8f z = {0.f,0.f,0.f,0.f,0.f,0.f,0.f,0.f};
  return z;
}

__device__ __forceinline__ v8f wmma16(v16h a, v16h b, v8f c) {
  // v_wmma_f32_16x16x32_f16  (8-arg form: neg_a, A, neg_b, B, c_mod, C, reuse_a, reuse_b)
  return __builtin_amdgcn_wmma_f32_16x16x32_f16(false, a, false, b, (short)0, c, false, false);
}

// A fragment: 16x32 f16 (M rows x K), row-major src with row stride ld (halves).
// Layout used (contiguous k, matches load_bt): lanes 0-15 -> M=lane, elems e -> K=e;
//             lanes 16-31 -> same M, K=16+e.
__device__ __forceinline__ v16h load_a(const _Float16* src, int ld) {
  int lane = threadIdx.x & 31;
  const _Float16* p = src + (lane & 15) * ld + ((lane >> 4) << 4);
  v8h lo = *(const v8h*)(p);
  v8h hi = *(const v8h*)(p + 8);
  v16h r;
  #pragma unroll
  for (int i = 0; i < 8; ++i) { r[i] = lo[i]; r[i + 8] = hi[i]; }
  return r;
}

// B fragment: 32x16 f16 (K x N) where logical B[k][n] = src[n*ld + k]
// (i.e. src holds the transpose row-major -> per-lane contiguous 32B load).
// Layout used (contiguous k, matches load_a): lanes 0-15 -> col=lane, elems e -> K=e; lanes 16-31 -> K=16+e.
__device__ __forceinline__ v16h load_bt(const _Float16* src, int ld) {
  int lane = threadIdx.x & 31;
  const _Float16* p = src + (lane & 15) * ld + ((lane >> 4) << 4);
  v8h lo = *(const v8h*)(p);
  v8h hi = *(const v8h*)(p + 8);
  v16h r;
  #pragma unroll
  for (int i = 0; i < 8; ++i) { r[i] = lo[i]; r[i + 8] = hi[i]; }
  return r;
}

// ---------------------------------------------------------------- conversions
__global__ __launch_bounds__(256)
void cvt_f32_to_f16(const float* __restrict__ in, _Float16* __restrict__ out, int n4) {
  int i = blockIdx.x * 256 + threadIdx.x;
  if (i >= n4) return;
  v4f f = *(const v4f*)(in + (size_t)i * 4);
  v4h h;
  #pragma unroll
  for (int r = 0; r < 4; ++r) h[r] = (_Float16)f[r];
  *(v4h*)(out + (size_t)i * 4) = h;
}

// ------------------------------------------- fused Linear + BN + LIF (4 t's)
// D[d,m] = sum_c W[d,c] * x[m,c];   A = W (16 d-rows x 32 c),  B = x^T view.
// Each wave: 16 m x 32 d tile, 4 time-step accumulators, LIF scan in registers.
template<typename OutT>
__global__ __launch_bounds__(256)
void gemm_bn_lif(const _Float16* __restrict__ xh,   // [T][MM][CC] f16
                 const _Float16* __restrict__ wh,   // [CC out][CC in] f16
                 const float* __restrict__ lb,      // [CC]
                 const float* __restrict__ bng, const float* __restrict__ bnb,
                 const float* __restrict__ bnm, const float* __restrict__ bnv,
                 float vth,
                 OutT* __restrict__ out)            // [T][MM][CC]
{
  int wid    = blockIdx.x * 8 + (threadIdx.x >> 5);
  int dstrip = wid & 15;               // 16 strips of 32 output cols
  int mtile  = wid >> 4;               // 512 tiles of 16 rows
  int m0 = mtile << 4;
  int d0 = dstrip << 5;

  v8f acc[TT][2];
  #pragma unroll
  for (int t = 0; t < TT; ++t) { acc[t][0] = zero8(); acc[t][1] = zero8(); }

  for (int kb = 0; kb < CC; kb += 32) {
    // prefetch next k-step's W lines into cache (global_prefetch_b8)
    if (kb + 32 < CC) {
      __builtin_prefetch(wh + (size_t)d0 * CC + kb + 32, 0, 0);
      __builtin_prefetch(wh + (size_t)(d0 + 16) * CC + kb + 32, 0, 0);
    }
    v16h a0 = load_a(wh + (size_t)d0 * CC + kb, CC);
    v16h a1 = load_a(wh + (size_t)(d0 + 16) * CC + kb, CC);
    #pragma unroll
    for (int t = 0; t < TT; ++t) {
      v16h b = load_bt(xh + (size_t)t * MC + (size_t)m0 * CC + kb, CC);
      acc[t][0] = wmma16(a0, b, acc[t][0]);
      acc[t][1] = wmma16(a1, b, acc[t][1]);
    }
  }

  int lane = threadIdx.x & 31;
  int m = m0 + (lane & 15);
  #pragma unroll
  for (int j = 0; j < 2; ++j) {
    int dbase = d0 + j * 16 + ((lane >> 4) << 3);   // 8 contiguous d per lane
    float sc[8], sh[8];
    #pragma unroll
    for (int r = 0; r < 8; ++r) {
      int d = dbase + r;
      float s = bng[d] * rsqrtf(bnv[d] + 1e-5f);
      sc[r] = s;
      sh[r] = (lb[d] - bnm[d]) * s + bnb[d];
    }
    float vm[8];
    #pragma unroll
    for (int r = 0; r < 8; ++r) vm[r] = 0.f;
    #pragma unroll
    for (int t = 0; t < TT; ++t) {
      float spk[8];
      #pragma unroll
      for (int r = 0; r < 8; ++r) {
        float y = acc[t][j][r] * sc[r] + sh[r];
        vm[r] = 0.5f * (vm[r] + y);                 // v += (x - v)/tau, tau=2
        float s = (vm[r] >= vth) ? 1.0f : 0.0f;
        vm[r] = (1.0f - s) * vm[r];                 // hard reset
        spk[r] = s;
      }
      size_t ofs = (size_t)t * MC + (size_t)m * CC + dbase;
      if constexpr (sizeof(OutT) == 2) {
        v8h pk;
        #pragma unroll
        for (int r = 0; r < 8; ++r) pk[r] = (_Float16)spk[r];
        *(v8h*)((_Float16*)out + ofs) = pk;
      } else {
        v4f p0, p1;
        #pragma unroll
        for (int r = 0; r < 4; ++r) { p0[r] = spk[r]; p1[r] = spk[r + 4]; }
        *(v4f*)((float*)out + ofs)     = p0;
        *(v4f*)((float*)out + ofs + 4) = p1;
      }
    }
  }
}

// ---------------------------------------------------------------- attention
// One workgroup per (t,b,h). O = (q k^T * 0.125) v with q,k,v binary f16
// spike tiles 256x64 (row stride CC in the spike tensors).
// k tile is DMA'd into LDS by the Tensor Data Mover (one descriptor per WG).
__global__ __launch_bounds__(256)
void attention(const _Float16* __restrict__ sq, const _Float16* __restrict__ sk,
               const _Float16* __restrict__ sv, float* __restrict__ o)
{
  __shared__ _Float16 vT[DD * NN];        // v transposed: vT[d][m], 32 KB
  __shared__ _Float16 kbuf[NN * DD];      // k tile compact [m][d], 32 KB (TDM dest)
  __shared__ _Float16 Sbuf[8][16 * 32];   // per-wave S chunk (f16), 8 KB

  int gid = blockIdx.x;                   // t*256 + b*8 + h
  int t = gid >> 8;
  int b = (gid >> 3) & 31;
  int h = gid & 7;
  size_t base = (size_t)t * MC + (size_t)b * NN * CC + (size_t)h * DD;
  const _Float16* q = sq + base;
  const _Float16* k = sk + base;
  const _Float16* v = sv + base;

#if __has_builtin(__builtin_amdgcn_tensor_load_to_lds)
  // --- TDM: DMA k tile [256 rows x 64 cols, row stride 512 halves] -> kbuf
  if (threadIdx.x < 32) {
    unsigned long long ga = (unsigned long long)(uintptr_t)k;
    unsigned int lds_off = (unsigned int)(uintptr_t)&kbuf[0];  // generic LDS addr[31:0] = LDS offset
    v4u g0;
    g0[0] = 1u;                                   // count=1, user descriptor
    g0[1] = lds_off;                              // lds_addr (bytes)
    g0[2] = (unsigned int)ga;                     // global_addr[31:0]
    g0[3] = (unsigned int)((ga >> 32) & 0x1FFFFFFu) | (2u << 30); // addr[56:32] | type=2
    v8i g1;
    g1[0] = (int)(1u << 16);                      // workgroup_mask=0, data_size=1 (2B)
    g1[1] = (int)(64u << 16);                     // tensor_dim0[15:0]=64 in bits 63:48
    g1[2] = (int)(256u << 16);                    // tensor_dim1[15:0]=256 in bits 95:80
    g1[3] = (int)(64u << 16);                     // tile_dim0=64 in bits 127:112
    g1[4] = 256;                                  // tile_dim1=256 (bits 143:128), tile_dim2=0
    g1[5] = 512;                                  // tensor_dim0_stride=512 (bits 207:160)
    g1[6] = 0;                                    // stride hi / tensor_dim1_stride lo
    g1[7] = 0;
    v4i g2 = {0,0,0,0};                           // 2D tensor: groups 2/3 unused
    v4i g3 = {0,0,0,0};
    v8i g4 = {0,0,0,0,0,0,0,0};                   // extra group (6-arg toolchain form)
    __builtin_amdgcn_tensor_load_to_lds(g0, g1, g2, g3, g4, 0);
  }
#else
  for (int i = threadIdx.x; i < NN * DD; i += 256) {
    int d2 = i & (DD - 1);
    int m2 = i >> 6;
    kbuf[m2 * DD + d2] = k[(size_t)m2 * CC + d2];
  }
#endif

  // cooperative transpose of v into LDS: vT[d][m] = v[m*CC + d]
  for (int i = threadIdx.x; i < DD * NN; i += 256) {
    int d2 = i & (DD - 1);
    int m2 = i >> 6;
    vT[d2 * NN + m2] = v[(size_t)m2 * CC + d2];
  }

#if __has_builtin(__builtin_amdgcn_tensor_load_to_lds)
  if (threadIdx.x < 32) {
#if __has_builtin(__builtin_amdgcn_s_wait_tensorcnt)
    __builtin_amdgcn_s_wait_tensorcnt(0);
#else
    asm volatile("s_wait_tensorcnt 0x0" ::: "memory");
#endif
  }
#endif
  __syncthreads();

  int wave = threadIdx.x >> 5;
  int lane = threadIdx.x & 31;
  _Float16* Sw = &Sbuf[wave][0];
  int rofs = (lane >> 4) << 3;
  int col  = lane & 15;

  #pragma unroll
  for (int nc = 0; nc < 2; ++nc) {
    int n0 = nc * 128 + wave * 16;
    const _Float16* qrow = q + (size_t)n0 * CC;

    v8f oacc[4];
    #pragma unroll
    for (int dt = 0; dt < 4; ++dt) oacc[dt] = zero8();

    for (int mcix = 0; mcix < 8; ++mcix) {
      int mb = mcix * 32;
      // S chunk [16 n x 32 m] = q[16x64] @ k^T   (k from LDS kbuf, compact ld=64)
      v8f s0 = zero8(), s1 = zero8();
      #pragma unroll
      for (int kb = 0; kb < DD; kb += 32) {
        v16h aq = load_a(qrow + kb, CC);
        v16h b0 = load_bt(kbuf + (size_t)(mb + 0)  * DD + kb, DD);
        v16h b1 = load_bt(kbuf + (size_t)(mb + 16) * DD + kb, DD);
        s0 = wmma16(aq, b0, s0);
        s1 = wmma16(aq, b1, s1);
      }
      // scale and spill to per-wave LDS chunk (f16, row-major 16x32)
      #pragma unroll
      for (int r = 0; r < 8; ++r) {
        Sw[(rofs + r) * 32 + col]      = (_Float16)(s0[r] * 0.125f);
        Sw[(rofs + r) * 32 + 16 + col] = (_Float16)(s1[r] * 0.125f);
      }
      asm volatile("s_wait_dscnt 0x0" ::: "memory");   // lanes' S visible to wave
      // O[16x64] += S[16x32] @ v_chunk[32x64]   (B from vT: B[m][d] = vT[d][m])
      v16h as = load_a(Sw, 32);
      #pragma unroll
      for (int dt = 0; dt < 4; ++dt) {
        v16h bv = load_bt(vT + (size_t)(dt * 16) * NN + mb, NN);
        oacc[dt] = wmma16(as, bv, oacc[dt]);
      }
      asm volatile("s_wait_dscnt 0x0" ::: "memory");   // finish reads before next chunk overwrites
    }

    #pragma unroll
    for (int dt = 0; dt < 4; ++dt) {
      int d = dt * 16 + col;
      #pragma unroll
      for (int r = 0; r < 8; ++r) {
        int n = n0 + rofs + r;
        o[base + (size_t)n * CC + d] = oacc[dt][r];
      }
    }
  }
}

// --------------------------------------------------------- LIF on attn output
__global__ __launch_bounds__(256)
void lif_o(const float* __restrict__ o, _Float16* __restrict__ s, float vth) {
  int i = blockIdx.x * 256 + threadIdx.x;      // over MM*CC
  float vm = 0.f;
  #pragma unroll
  for (int t = 0; t < TT; ++t) {
    float y = o[(size_t)t * MC + i];
    vm = 0.5f * (vm + y);
    float sp = (vm >= vth) ? 1.0f : 0.0f;
    vm = (1.0f - sp) * vm;
    s[(size_t)t * MC + i] = (_Float16)sp;
  }
}

// ------------------------------------------------------------------- launch
extern "C" void kernel_launch(void* const* d_in, const int* in_sizes, int n_in,
                              void* d_out, int out_size, void* d_ws, size_t ws_size,
                              hipStream_t stream) {
  const float* x   = (const float*)d_in[0];  // [T,B,N,C]
  const float* lw  = (const float*)d_in[1];  // [4,C,C]
  const float* lbi = (const float*)d_in[2];  // [4,C]
  const float* bg  = (const float*)d_in[3];
  const float* bb  = (const float*)d_in[4];
  const float* bm  = (const float*)d_in[5];
  const float* bv  = (const float*)d_in[6];
  float* out = (float*)d_out;

  char* ws = (char*)d_ws;
  _Float16* xh  = (_Float16*)ws;  ws += (size_t)TT * MC * sizeof(_Float16);
  _Float16* wh  = (_Float16*)ws;  ws += (size_t)4 * CC * CC * sizeof(_Float16);
  _Float16* sq  = (_Float16*)ws;  ws += (size_t)TT * MC * sizeof(_Float16);
  _Float16* sk  = (_Float16*)ws;  ws += (size_t)TT * MC * sizeof(_Float16);
  _Float16* sv  = (_Float16*)ws;  ws += (size_t)TT * MC * sizeof(_Float16);
  _Float16* osp = (_Float16*)ws;  ws += (size_t)TT * MC * sizeof(_Float16);
  float*    ob  = (float*)ws;     // [T][MM][CC] f32

  // f32 -> f16 conversions
  {
    int n4 = (TT * MC) / 4;
    cvt_f32_to_f16<<<n4 / 256, 256, 0, stream>>>(x, xh, n4);
    int w4 = (4 * CC * CC) / 4;
    cvt_f32_to_f16<<<w4 / 256, 256, 0, stream>>>(lw, wh, w4);
  }

  const int GEMM_BLOCKS = (MM / 16) * (CC / 32) / 8;   // 1024

  // q / k / v branches (spike output f16)
  gemm_bn_lif<_Float16><<<GEMM_BLOCKS, 256, 0, stream>>>(
      xh, wh + 0 * CC * CC, lbi + 0 * CC, bg + 0 * CC, bb + 0 * CC,
      bm + 0 * CC, bv + 0 * CC, 1.0f, sq);
  gemm_bn_lif<_Float16><<<GEMM_BLOCKS, 256, 0, stream>>>(
      xh, wh + 1 * CC * CC, lbi + 1 * CC, bg + 1 * CC, bb + 1 * CC,
      bm + 1 * CC, bv + 1 * CC, 1.0f, sk);
  gemm_bn_lif<_Float16><<<GEMM_BLOCKS, 256, 0, stream>>>(
      xh, wh + 2 * CC * CC, lbi + 2 * CC, bg + 2 * CC, bb + 2 * CC,
      bm + 2 * CC, bv + 2 * CC, 1.0f, sv);

  // attention: one block per (t, b, h)
  attention<<<TT * BB * HH, 256, 0, stream>>>(sq, sk, sv, ob);

  // attn LIF (threshold 0.5) -> spikes f16
  lif_o<<<MC / 256, 256, 0, stream>>>(ob, osp, 0.5f);

  // projection branch, spikes written as f32 directly to d_out
  gemm_bn_lif<float><<<GEMM_BLOCKS, 256, 0, stream>>>(
      osp, wh + 3 * CC * CC, lbi + 3 * CC, bg + 3 * CC, bb + 3 * CC,
      bm + 3 * CC, bv + 3 * CC, 1.0f, out);
}
